// Attention_10179072491608
// MI455X (gfx1250) — matrix operand order repaired and checked by reading, not compiled, not run on hardware
//
#include <hip/hip_runtime.h>
#include <hip/hip_bf16.h>
#include <math.h>

// ---------------------------------------------------------------------------
// Problem dims (compile-time, from the reference)
// ---------------------------------------------------------------------------
constexpr int kB = 8, kT = 512, kE = 1024, kL = 8, kD = 512;
constexpr int kBT = kB * kT;   // 4096
constexpr int kLD = kL * kD;   // 4096
constexpr float kEPS = 1e-5f;

typedef __bf16 bf16;
typedef __attribute__((ext_vector_type(16))) __bf16 v16bf;
typedef __attribute__((ext_vector_type(8)))  float  v8f;
typedef __attribute__((ext_vector_type(4)))  unsigned int v4u;
typedef __attribute__((ext_vector_type(8)))  int v8i;
typedef __attribute__((ext_vector_type(4)))  int v4i;

union FragU { v16bf v; float4 q[2]; };

// ---------------------------------------------------------------------------
// WMMA helpers: D = A(16x32 bf16) x B(32x16 bf16) + C(16x16 f32)
// A: lane(l&15)=M row; elems 0..7 = K[(l>>4)*8..], elems 8..15 = K[16+(l>>4)*8..]
// B: lane(l&15)=N col; elems 0..7 = K[(l>>4)*8..], elems 8..15 = K[16+(l>>4)*8..] (weights transposed [N][K])
// C: lane(l&15)=N col; elem r = M row 8*(l>>4)+r
// ---------------------------------------------------------------------------
__device__ __forceinline__ v8f wmma_bf16(v16bf a, v16bf b, v8f c) {
  return __builtin_amdgcn_wmma_f32_16x16x32_bf16(false, a, false, b, (short)0, c,
                                                 false, false);
}

__device__ __forceinline__ v16bf load_a_frag(const bf16* __restrict__ A, int lda,
                                             int m0, int k0) {
  int lane = threadIdx.x & 31;
  int m  = m0 + (lane & 15);
  int kh = (lane >> 4) << 3;                      // 0 or 8
  const float4* p = reinterpret_cast<const float4*>(A + (size_t)m * lda + k0 + kh);
  FragU f; f.q[0] = p[0]; f.q[1] = p[2];
  return f.v;
}

// A fragment from an 8x512 LDS tile, rows 8..15 implicitly zero (M padded to 16)
__device__ __forceinline__ v16bf load_a_frag_lds8(const bf16* __restrict__ A, int k0) {
  int lane = threadIdx.x & 31;
  int m  = lane & 15;
  int kh = (lane >> 4) << 3;
  FragU f;
  f.q[0] = make_float4(0.f, 0.f, 0.f, 0.f);
  f.q[1] = f.q[0];
  if (m < 8) {
    const float4* p = reinterpret_cast<const float4*>(A + m * kD + k0 + kh);
    f.q[0] = p[0]; f.q[1] = p[2];
  }
  return f.v;
}

__device__ __forceinline__ v16bf load_b_frag(const bf16* __restrict__ WT, int ldk,
                                             int n0, int k0) {
  int lane = threadIdx.x & 31;
  int n  = n0 + (lane & 15);
  int kb = k0 + ((lane >> 4) << 3);               // 0 or 8
  const float4* p = reinterpret_cast<const float4*>(WT + (size_t)n * ldk + kb);
  FragU f; f.q[0] = p[0]; f.q[1] = p[2];
  return f.v;
}

// Pipelined 32x64 wave tile: double-buffered fragments so WMMAs overlap loads.
template <int K>
__device__ __forceinline__ void gemm_core_32x64(const bf16* __restrict__ A, int lda,
                                                const bf16* __restrict__ WT, int ldk,
                                                int m0, int n0, v8f acc[2][4]) {
  v16bf a0 = load_a_frag(A, lda, m0, 0);
  v16bf a1 = load_a_frag(A, lda, m0 + 16, 0);
  v16bf b0 = load_b_frag(WT, ldk, n0, 0);
  v16bf b1 = load_b_frag(WT, ldk, n0 + 16, 0);
  v16bf b2 = load_b_frag(WT, ldk, n0 + 32, 0);
  v16bf b3 = load_b_frag(WT, ldk, n0 + 48, 0);
  for (int k0 = 0; k0 < K; k0 += 32) {
    int kn = k0 + 32;
    v16bf na0, na1, nb0, nb1, nb2, nb3;
    if (kn < K) {
      na0 = load_a_frag(A, lda, m0, kn);
      na1 = load_a_frag(A, lda, m0 + 16, kn);
      nb0 = load_b_frag(WT, ldk, n0, kn);
      nb1 = load_b_frag(WT, ldk, n0 + 16, kn);
      nb2 = load_b_frag(WT, ldk, n0 + 32, kn);
      nb3 = load_b_frag(WT, ldk, n0 + 48, kn);
      __builtin_prefetch(WT + (size_t)(n0 + (threadIdx.x & 15)) * ldk + kn + 32, 0, 1);
    }
    acc[0][0] = wmma_bf16(a0, b0, acc[0][0]);
    acc[0][1] = wmma_bf16(a0, b1, acc[0][1]);
    acc[0][2] = wmma_bf16(a0, b2, acc[0][2]);
    acc[0][3] = wmma_bf16(a0, b3, acc[0][3]);
    acc[1][0] = wmma_bf16(a1, b0, acc[1][0]);
    acc[1][1] = wmma_bf16(a1, b1, acc[1][1]);
    acc[1][2] = wmma_bf16(a1, b2, acc[1][2]);
    acc[1][3] = wmma_bf16(a1, b3, acc[1][3]);
    if (kn < K) { a0 = na0; a1 = na1; b0 = nb0; b1 = nb1; b2 = nb2; b3 = nb3; }
  }
}

__device__ __forceinline__ void zero_acc(v8f acc[2][4]) {
  #pragma unroll
  for (int i = 0; i < 2; ++i)
    #pragma unroll
    for (int j = 0; j < 4; ++j)
      acc[i][j] = (v8f){0.f,0.f,0.f,0.f,0.f,0.f,0.f,0.f};
}

__device__ __forceinline__ float block_sum_256(float v, float* s_red) {
  int w = threadIdx.x >> 5, lane = threadIdx.x & 31;
  #pragma unroll
  for (int off = 16; off; off >>= 1) v += __shfl_xor(v, off, 32);
  __syncthreads();
  if (lane == 0) s_red[w] = v;
  __syncthreads();
  float s = 0.f;
  #pragma unroll
  for (int i = 0; i < 8; ++i) s += s_red[i];
  return s;
}

// ---------------------------------------------------------------------------
// Tensor Data Mover: 2D tile (8 rows x 512 f32, row stride = kT*kLD elems)
// from global `gptr` into LDS byte offset `lds_off`.  D# per CDNA5 ISA §8.
// This toolchain exposes the 6-arg builtin (g0, g1, g2, g3, g4, cpol).
// ---------------------------------------------------------------------------
__device__ __forceinline__ void tdm_load_tok(const float* gptr, unsigned lds_off) {
  unsigned long long ga = (unsigned long long)(uintptr_t)gptr;
  v4u g0;
  g0[0] = 1u;                                        // count=1 (valid user D#)
  g0[1] = lds_off;                                   // lds_addr (bytes)
  g0[2] = (unsigned)(ga & 0xffffffffu);              // global_addr[31:0]
  g0[3] = (unsigned)((ga >> 32) & 0x01ffffffu)       // global_addr[56:32]
        | (2u << 30);                                // type=2 ("image")
  unsigned w0 = (2u << 16);                          // data_size=2 -> 4 bytes
  unsigned w1 = (512u << 16);                        // tensor_dim0[15:0] @bit48
  unsigned w2 = (8u << 16);                          // dim0 hi=0 | tensor_dim1 lo
  unsigned w3 = (512u << 16);                        // tile_dim0 @bit112
  unsigned w4 = 8u;                                  // tile_dim1 @bit128
  unsigned w5 = (unsigned)((size_t)kT * kLD);        // tensor_dim0_stride lo
  v8i g1 = { (int)w0, (int)w1, (int)w2, (int)w3, (int)w4, (int)w5, 0, 0 };
  v4i gz4 = { 0, 0, 0, 0 };
  v8i gz8 = { 0, 0, 0, 0, 0, 0, 0, 0 };
  __builtin_amdgcn_tensor_load_to_lds(g0, g1, gz4, gz4, gz8, 0);
}

// ---------------------------------------------------------------------------
// K-prep kernels
// ---------------------------------------------------------------------------
__global__ __launch_bounds__(256) void k_cvt_bf16(const float* __restrict__ s,
                                                  bf16* __restrict__ d, int n) {
  int i = blockIdx.x * 256 + threadIdx.x;
  if (i < n) d[i] = (bf16)s[i];
}

__global__ __launch_bounds__(256) void k_transpose_bf16(const float* __restrict__ src,
                                                        bf16* __restrict__ dst,
                                                        int R, int C) {
  __shared__ bf16 tile[32][33];
  const float* S = src + (size_t)blockIdx.z * R * C;
  bf16* Dp = dst + (size_t)blockIdx.z * R * C;
  int r0 = blockIdx.y * 32, c0 = blockIdx.x * 32;
  int tx = threadIdx.x & 31, ty = threadIdx.x >> 5;
  #pragma unroll
  for (int i = 0; i < 4; ++i)
    tile[ty + 8 * i][tx] = (bf16)S[(size_t)(r0 + ty + 8 * i) * C + c0 + tx];
  __syncthreads();
  #pragma unroll
  for (int i = 0; i < 4; ++i)
    Dp[(size_t)(c0 + ty + 8 * i) * R + r0 + tx] = tile[tx][ty + 8 * i];
}

__global__ __launch_bounds__(256) void k_rowsum3(const float* __restrict__ a,
                                                 const float* __restrict__ b,
                                                 const float* __restrict__ c,
                                                 float* __restrict__ sums) {
  __shared__ float s_red[8];
  int row = blockIdx.x;
  const float* ptrs[3] = {a, b, c};
  for (int t = 0; t < 3; ++t) {
    const float* p = ptrs[t] + (size_t)row * kD;
    float v = p[threadIdx.x] + p[threadIdx.x + 256];
    float s = block_sum_256(v, s_red);
    if (threadIdx.x == 0) sums[t * kLD + row] = s;
  }
}

// ---------------------------------------------------------------------------
// tokens[4096][4096] = emb_bf16[4096][1024] @ e2s_wT + e2s_b   (f32 out)
// ---------------------------------------------------------------------------
__global__ __launch_bounds__(256) void k_e2s(const bf16* __restrict__ A,
                                             const bf16* __restrict__ WT,
                                             const float* __restrict__ bias,
                                             float* __restrict__ C) {
  int w = threadIdx.x >> 5, lane = threadIdx.x & 31;
  int m0 = blockIdx.x * 64 + (w >> 2) * 32;
  int n0 = blockIdx.y * 256 + (w & 3) * 64;
  v8f acc[2][4];
  zero_acc(acc);
  gemm_core_32x64<kE>(A, kE, WT, kE, m0, n0, acc);
  #pragma unroll
  for (int mi = 0; mi < 2; ++mi) {
    int mrow = m0 + 16 * mi + ((lane >> 4) << 3);
    #pragma unroll
    for (int j = 0; j < 4; ++j) {
      int n = n0 + 16 * j + (lane & 15);
      float bn = bias[n];
      #pragma unroll
      for (int r = 0; r < 8; ++r)
        C[(size_t)(mrow + r) * kLD + n] = acc[mi][j][r] + bn;
    }
  }
}

// ---------------------------------------------------------------------------
// Per-layer persistent scan.  One WG per layer, 512 threads = 16 waves.
// Dynamic LDS layout (byte offsets):
//   0      s_state f32 [8][512]   (16384)
//   16384  s_hn    bf16 [8][512]  ( 8192)
//   24576  s_proj  f32 [8][1024]  (32768)
//   57344  s_part  f32 [16]       (   64)
//   57600  s_tok   f32 [2][8][512](32768)   <- TDM double buffer
// total 90368 bytes (dynamic; > 64KB static limit, fine in a 320KB WGP)
// ---------------------------------------------------------------------------
constexpr unsigned kScanLds = 90368;

__global__ __launch_bounds__(512) void k_scan(const float* __restrict__ tokens,
                                              const bf16* __restrict__ WGT,
                                              const float* __restrict__ glu_b,
                                              const float* __restrict__ sums,
                                              const float* __restrict__ lng,
                                              const float* __restrict__ lnb,
                                              const float* __restrict__ lsg,
                                              const float* __restrict__ lsb,
                                              float* __restrict__ attn) {
  extern __shared__ char smem[];
  float* s_state = (float*)(smem);
  bf16*  s_hn    = (bf16*)(smem + 16384);
  float* s_proj  = (float*)(smem + 24576);
  float* s_part  = (float*)(smem + 57344);
  float* s_tok   = (float*)(smem + 57600);

  int l = blockIdx.x;
  int tid = threadIdx.x;
  int w = tid >> 5, lane = tid & 31;
  int b = tid >> 6, q = tid & 63;                    // row b, 64 threads per row
  const bf16* WT = WGT + (size_t)l * 1024 * 512;
  const float* gb = glu_b + l * 1024;
  const float* sc = sums + (size_t)l * kD;
  const float* ii = sums + (size_t)(kL + l) * kD;
  const float* os = sums + (size_t)(2 * kL + l) * kD;

  for (int i = tid; i < 8 * 512; i += 512) s_state[i] = 0.f;

  // TDM prefetch of the t=0 token tile into buffer 0
  if (w == 0) tdm_load_tok(tokens + (size_t)0 * kLD + l * kD, 57600u);
  __syncthreads();

  for (int t = 0; t < kT; ++t) {
    // issue TDM for t+1, then wait until the t-th tile has landed
    if (w == 0) {
      if (t + 1 < kT) {
        tdm_load_tok(tokens + (size_t)(t + 1) * kLD + l * kD,
                     57600u + (unsigned)(((t + 1) & 1) * 16384));
        __builtin_amdgcn_s_wait_tensorcnt(1);
      } else {
        __builtin_amdgcn_s_wait_tensorcnt(0);
      }
    }
    __syncthreads();
    const float* tok = s_tok + (t & 1) * 4096 + b * 512;

    // --- h = state*sc + tok*ii ; LN(ln_glu) -> bf16 s_hn -----------------
    float hv[8]; float part = 0.f;
    #pragma unroll
    for (int i = 0; i < 8; ++i) {
      int d = q + 64 * i;
      float h = s_state[b * 512 + d] * sc[d] + tok[d] * ii[d];
      hv[i] = h; part += h;
    }
    #pragma unroll
    for (int off = 16; off; off >>= 1) part += __shfl_xor(part, off, 32);
    if (lane == 0) s_part[w] = part;
    __syncthreads();
    float mu = (s_part[2 * b] + s_part[2 * b + 1]) * (1.f / 512.f);
    part = 0.f;
    #pragma unroll
    for (int i = 0; i < 8; ++i) { float dd = hv[i] - mu; part += dd * dd; }
    #pragma unroll
    for (int off = 16; off; off >>= 1) part += __shfl_xor(part, off, 32);
    __syncthreads();
    if (lane == 0) s_part[w] = part;
    __syncthreads();
    float rs = rsqrtf((s_part[2 * b] + s_part[2 * b + 1]) * (1.f / 512.f) + kEPS);
    #pragma unroll
    for (int i = 0; i < 8; ++i) {
      int d = q + 64 * i;
      s_hn[b * 512 + d] = (bf16)((hv[i] - mu) * rs * lng[d] + lnb[d]);
    }
    __syncthreads();

    // --- proj[16x1024] = hn x glu_w[l], wave w owns cols [64w,64w+64) -----
    {
      int n0 = w * 64;
      v8f acc[4];
      #pragma unroll
      for (int j = 0; j < 4; ++j) acc[j] = (v8f){0.f,0.f,0.f,0.f,0.f,0.f,0.f,0.f};
      v16bf b0 = load_b_frag(WT, kD, n0, 0);
      v16bf b1 = load_b_frag(WT, kD, n0 + 16, 0);
      v16bf b2 = load_b_frag(WT, kD, n0 + 32, 0);
      v16bf b3 = load_b_frag(WT, kD, n0 + 48, 0);
      for (int k0 = 0; k0 < kD; k0 += 32) {
        int kn = k0 + 32;
        v16bf nb0, nb1, nb2, nb3;
        if (kn < kD) {
          nb0 = load_b_frag(WT, kD, n0, kn);
          nb1 = load_b_frag(WT, kD, n0 + 16, kn);
          nb2 = load_b_frag(WT, kD, n0 + 32, kn);
          nb3 = load_b_frag(WT, kD, n0 + 48, kn);
        }
        v16bf a = load_a_frag_lds8(s_hn, k0);
        acc[0] = wmma_bf16(a, b0, acc[0]);
        acc[1] = wmma_bf16(a, b1, acc[1]);
        acc[2] = wmma_bf16(a, b2, acc[2]);
        acc[3] = wmma_bf16(a, b3, acc[3]);
        if (kn < kD) { b0 = nb0; b1 = nb1; b2 = nb2; b3 = nb3; }
      }
      if (lane < 16) {                               // rows 0..7 only
        #pragma unroll
        for (int j = 0; j < 4; ++j) {
          int n = n0 + 16 * j + lane;
          float bn = gb[n];
          #pragma unroll
          for (int r = 0; r < 8; ++r) s_proj[r * 1024 + n] = acc[j][r] + bn;
        }
      }
    }
    __syncthreads();

    // --- GLU + LN(ln_state) -> new state ; attn = state * os_sum ---------
    float vv[8]; part = 0.f;
    #pragma unroll
    for (int i = 0; i < 8; ++i) {
      int d = q + 64 * i;
      float a = s_proj[b * 1024 + d];
      float g = s_proj[b * 1024 + 512 + d];
      float val = a / (1.f + __expf(-g));
      vv[i] = val; part += val;
    }
    #pragma unroll
    for (int off = 16; off; off >>= 1) part += __shfl_xor(part, off, 32);
    __syncthreads();
    if (lane == 0) s_part[w] = part;
    __syncthreads();
    float mu2 = (s_part[2 * b] + s_part[2 * b + 1]) * (1.f / 512.f);
    part = 0.f;
    #pragma unroll
    for (int i = 0; i < 8; ++i) { float dd = vv[i] - mu2; part += dd * dd; }
    #pragma unroll
    for (int off = 16; off; off >>= 1) part += __shfl_xor(part, off, 32);
    __syncthreads();
    if (lane == 0) s_part[w] = part;
    __syncthreads();
    float rs2 = rsqrtf((s_part[2 * b] + s_part[2 * b + 1]) * (1.f / 512.f) + kEPS);
    float* arow = attn + ((size_t)(b * kT + t) * kL + l) * kD;
    #pragma unroll
    for (int i = 0; i < 8; ++i) {
      int d = q + 64 * i;
      float ns = (vv[i] - mu2) * rs2 * lsg[d] + lsb[d];
      s_state[b * 512 + d] = ns;
      arow[d] = ns * os[d];
    }
    __syncthreads();
  }
}

// ---------------------------------------------------------------------------
// LN per 512-row of attn -> bf16 yn (wave per row); rows = BT*L, l = row % L
// ---------------------------------------------------------------------------
__global__ __launch_bounds__(256) void k_ln_rows(const float* __restrict__ X,
                                                 const float* __restrict__ g,
                                                 const float* __restrict__ bb,
                                                 bf16* __restrict__ Y) {
  int row = blockIdx.x * 8 + (threadIdx.x >> 5);
  int lane = threadIdx.x & 31;
  int l = row & (kL - 1);
  const float* x = X + (size_t)row * kD;
  float v[16]; float p = 0.f;
  #pragma unroll
  for (int i = 0; i < 16; ++i) { v[i] = x[lane + 32 * i]; p += v[i]; }
  #pragma unroll
  for (int off = 16; off; off >>= 1) p += __shfl_xor(p, off, 32);
  float mu = p * (1.f / 512.f);
  float q = 0.f;
  #pragma unroll
  for (int i = 0; i < 16; ++i) { float d = v[i] - mu; q += d * d; }
  #pragma unroll
  for (int off = 16; off; off >>= 1) q += __shfl_xor(q, off, 32);
  float rs = rsqrtf(q * (1.f / 512.f) + kEPS);
  bf16* y = Y + (size_t)row * kD;
  #pragma unroll
  for (int i = 0; i < 16; ++i) {
    int d = lane + 32 * i;
    y[d] = (bf16)((v[i] - mu) * rs * g[l * kD + d] + bb[l * kD + d]);
  }
}

// ---------------------------------------------------------------------------
// h1 = GELU(yn @ ff_w1[l] + b1)  [per layer z]
// ---------------------------------------------------------------------------
__global__ __launch_bounds__(256) void k_ff1(const bf16* __restrict__ Yn,
                                             const bf16* __restrict__ W1T,
                                             const float* __restrict__ b1,
                                             bf16* __restrict__ H1) {
  int l = blockIdx.z;
  const bf16* A  = Yn + l * kD;
  const bf16* WT = W1T + (size_t)l * kD * kD;
  const float* bias = b1 + l * kD;
  bf16* Cp = H1 + l * kD;
  int w = threadIdx.x >> 5, lane = threadIdx.x & 31;
  int m0 = blockIdx.x * 64 + (w >> 2) * 32;
  int n0 = blockIdx.y * 256 + (w & 3) * 64;
  v8f acc[2][4];
  zero_acc(acc);
  gemm_core_32x64<kD>(A, kLD, WT, kD, m0, n0, acc);
  #pragma unroll
  for (int mi = 0; mi < 2; ++mi) {
    int mrow = m0 + 16 * mi + ((lane >> 4) << 3);
    #pragma unroll
    for (int j = 0; j < 4; ++j) {
      int n = n0 + 16 * j + (lane & 15);
      float bn = bias[n];
      #pragma unroll
      for (int r = 0; r < 8; ++r) {
        float x = acc[mi][j][r] + bn;
        float ge = 0.5f * x * (1.f + erff(x * 0.70710678118f));  // exact GELU
        Cp[(size_t)(mrow + r) * kLD + n] = (bf16)ge;
      }
    }
  }
}

// ---------------------------------------------------------------------------
// lo = h1 @ ff_w2[l] + b2  [per layer z]  -> f32 [BT][L][E]
// ---------------------------------------------------------------------------
__global__ __launch_bounds__(256) void k_ff2(const bf16* __restrict__ H1,
                                             const bf16* __restrict__ W2T,
                                             const float* __restrict__ b2,
                                             float* __restrict__ LO) {
  int l = blockIdx.z;
  const bf16* A  = H1 + l * kD;
  const bf16* WT = W2T + (size_t)l * kE * kD;
  const float* bias = b2 + l * kE;
  float* Cp = LO + (size_t)l * kE;
  int w = threadIdx.x >> 5, lane = threadIdx.x & 31;
  int m0 = blockIdx.x * 64 + (w >> 2) * 32;
  int n0 = blockIdx.y * 256 + (w & 3) * 64;
  v8f acc[2][4];
  zero_acc(acc);
  gemm_core_32x64<kD>(A, kLD, WT, kD, m0, n0, acc);
  #pragma unroll
  for (int mi = 0; mi < 2; ++mi) {
    int mrow = m0 + 16 * mi + ((lane >> 4) << 3);
    #pragma unroll
    for (int j = 0; j < 4; ++j) {
      int n = n0 + 16 * j + (lane & 15);
      float bn = bias[n];
      #pragma unroll
      for (int r = 0; r < 8; ++r)
        Cp[(size_t)(mrow + r) * (kL * kE) + n] = acc[mi][j][r] + bn;
    }
  }
}

// ---------------------------------------------------------------------------
// Per-token: scores = softmax_l(lo . attn_w + attn_b); out = LN(sum_l s_l*lo_l)
// ---------------------------------------------------------------------------
__global__ __launch_bounds__(256) void k_final(const float* __restrict__ lo,
                                               const float* __restrict__ aw,
                                               const float* __restrict__ ab,
                                               const float* __restrict__ gE,
                                               const float* __restrict__ bE,
                                               float* __restrict__ out) {
  __shared__ float s_red[8];
  __shared__ float s_sc[8];
  int bt = blockIdx.x, tid = threadIdx.x;
  const float* base = lo + (size_t)bt * (kL * kE);
  for (int l = 0; l < kL; ++l) {
    float p = 0.f;
    #pragma unroll
    for (int i = 0; i < 4; ++i) {
      int e = tid + 256 * i;
      p += base[(size_t)l * kE + e] * aw[e];
    }
    float s = block_sum_256(p, s_red);
    if (tid == 0) s_sc[l] = s + ab[0];
  }
  __syncthreads();
  if (tid == 0) {
    float m = s_sc[0];
    #pragma unroll
    for (int l = 1; l < kL; ++l) m = fmaxf(m, s_sc[l]);
    float ssum = 0.f, ex[kL];
    #pragma unroll
    for (int l = 0; l < kL; ++l) { ex[l] = __expf(s_sc[l] - m); ssum += ex[l]; }
    #pragma unroll
    for (int l = 0; l < kL; ++l) s_sc[l] = ex[l] / ssum;
  }
  __syncthreads();
  float wv[4];
  #pragma unroll
  for (int i = 0; i < 4; ++i) {
    int e = tid + 256 * i;
    float acc = 0.f;
    #pragma unroll
    for (int l = 0; l < kL; ++l) acc += s_sc[l] * base[(size_t)l * kE + e];
    wv[i] = acc;
  }
  float p = wv[0] + wv[1] + wv[2] + wv[3];
  float mu = block_sum_256(p, s_red) * (1.f / (float)kE);
  float qv = 0.f;
  #pragma unroll
  for (int i = 0; i < 4; ++i) { float d = wv[i] - mu; qv += d * d; }
  float rs = rsqrtf(block_sum_256(qv, s_red) * (1.f / (float)kE) + kEPS);
  #pragma unroll
  for (int i = 0; i < 4; ++i) {
    int e = tid + 256 * i;
    out[(size_t)bt * kE + e] = (wv[i] - mu) * rs * gE[e] + bE[e];
  }
}

// ---------------------------------------------------------------------------
// Host launcher
// ---------------------------------------------------------------------------
extern "C" void kernel_launch(void* const* d_in, const int* in_sizes, int n_in,
                              void* d_out, int out_size, void* d_ws, size_t ws_size,
                              hipStream_t stream) {
  const float* emb    = (const float*)d_in[0];
  const float* sctrl  = (const float*)d_in[1];
  const float* iinfl  = (const float*)d_in[2];
  const float* oshap  = (const float*)d_in[3];
  const float* glu_w  = (const float*)d_in[4];
  const float* glu_b  = (const float*)d_in[5];
  const float* e2s_w  = (const float*)d_in[6];
  const float* e2s_b  = (const float*)d_in[7];
  const float* lngg   = (const float*)d_in[8];
  const float* lngb   = (const float*)d_in[9];
  const float* lnsg   = (const float*)d_in[10];
  const float* lnsb   = (const float*)d_in[11];
  const float* lneg   = (const float*)d_in[12];
  const float* lneb   = (const float*)d_in[13];
  const float* ffln_g = (const float*)d_in[14];
  const float* ffln_b = (const float*)d_in[15];
  const float* ff_w1  = (const float*)d_in[16];
  const float* ff_b1  = (const float*)d_in[17];
  const float* ff_w2  = (const float*)d_in[18];
  const float* ff_b2  = (const float*)d_in[19];
  const float* attn_w = (const float*)d_in[20];
  const float* attn_b = (const float*)d_in[21];
  float* out = (float*)d_out;

  char* ws = (char*)d_ws;
  size_t off = 0;
  auto alloc = [&](size_t bytes) -> void* {
    off = (off + 255) & ~(size_t)255;
    void* p = ws + off;
    off += bytes;
    return p;
  };
  bf16* emb_bf  = (bf16*)alloc((size_t)kBT * kE * 2);
  bf16* wt_e2s  = (bf16*)alloc((size_t)kLD * kE * 2);
  bf16* wt_glu  = (bf16*)alloc((size_t)kL * 1024 * 512 * 2);
  bf16* wt_ff1  = (bf16*)alloc((size_t)kL * 512 * 512 * 2);
  bf16* wt_ff2  = (bf16*)alloc((size_t)kL * 1024 * 512 * 2);
  float* sums   = (float*)alloc((size_t)3 * kLD * 4);
  float* tokens = (float*)alloc((size_t)kBT * kLD * 4);
  float* attn   = (float*)alloc((size_t)kBT * kLD * 4);
  float* lo     = (float*)alloc((size_t)kBT * kL * kE * 4);
  bf16* yn = (bf16*)tokens;                      // reuse: tokens dead after scan
  bf16* h1 = yn + (size_t)kBT * kLD;
  (void)ws_size; (void)n_in; (void)in_sizes; (void)out_size;

  k_cvt_bf16<<<(kBT * kE + 255) / 256, 256, 0, stream>>>(emb, emb_bf, kBT * kE);
  k_transpose_bf16<<<dim3(kLD / 32, kE / 32, 1),    256, 0, stream>>>(e2s_w, wt_e2s, kE, kLD);
  k_transpose_bf16<<<dim3(1024 / 32, 512 / 32, kL), 256, 0, stream>>>(glu_w, wt_glu, 512, 1024);
  k_transpose_bf16<<<dim3(512 / 32, 512 / 32, kL),  256, 0, stream>>>(ff_w1, wt_ff1, 512, 512);
  k_transpose_bf16<<<dim3(1024 / 32, 512 / 32, kL), 256, 0, stream>>>(ff_w2, wt_ff2, 512, 1024);
  k_rowsum3<<<kLD, 256, 0, stream>>>(sctrl, iinfl, oshap, sums);

  k_e2s<<<dim3(kBT / 64, kLD / 256), 256, 0, stream>>>(emb_bf, wt_e2s, e2s_b, tokens);

  k_scan<<<kL, 512, kScanLds, stream>>>(tokens, wt_glu, glu_b, sums,
                                        lngg, lngb, lnsg, lnsb, attn);

  k_ln_rows<<<(kBT * kL) / 8, 256, 0, stream>>>(attn, ffln_g, ffln_b, yn);
  k_ff1<<<dim3(kBT / 64, 512 / 256, kL),  256, 0, stream>>>(yn, wt_ff1, ff_b1, h1);
  k_ff2<<<dim3(kBT / 64, 1024 / 256, kL), 256, 0, stream>>>(h1, wt_ff2, ff_b2, lo);

  k_final<<<kBT, 256, 0, stream>>>(lo, attn_w, attn_b, lneg, lneb, out);
}
